// VanillaNeRFModel_10814727652056
// MI455X (gfx1250) — compile-verified, matrix operand order checked by reading, not run on hardware
//
#include <hip/hip_runtime.h>
#include <hip/hip_bf16.h>
#include <math.h>

typedef _Float16 half_t;
typedef __attribute__((ext_vector_type(16))) _Float16 v16h;
typedef __attribute__((ext_vector_type(8)))  float    v8f;

#define NRAYS 512
#define NS_C  64
#define NS_F  192   // 64 coarse + 128 importance, sorted
#define PC    (NRAYS * NS_C)   // 32768
#define PF    (NRAYS * NS_F)   // 98304

// ---------------------------------------------------------------------------
// Weight prep: fp32 [K,N] row-major -> f16 transposed [N, Kpad] (zero padded)
// ---------------------------------------------------------------------------
__global__ void prep_wt(const float* __restrict__ Wsrc, half_t* __restrict__ WT,
                        int K, int N, int Kpad) {
  int idx = blockIdx.x * blockDim.x + threadIdx.x;
  if (idx >= N * Kpad) return;
  int n = idx / Kpad, k = idx - n * Kpad;
  float v = (k < K) ? Wsrc[(size_t)k * N + n] : 0.0f;
  WT[idx] = (half_t)v;
}

// ---------------------------------------------------------------------------
// Per-ray prep: viewdir embedding (27 -> pad 32), |d|, coarse z_vals
// ---------------------------------------------------------------------------
__global__ void ray_prep(const float* __restrict__ rd,
                         const float* __restrict__ nearp,
                         const float* __restrict__ farp,
                         half_t* __restrict__ dirs_e,
                         float* __restrict__ dnorm,
                         float* __restrict__ zc) {
  int r = blockIdx.x * blockDim.x + threadIdx.x;
  if (r >= NRAYS) return;
  float dx = rd[r * 3 + 0], dy = rd[r * 3 + 1], dz = rd[r * 3 + 2];
  float nrm = sqrtf(dx * dx + dy * dy + dz * dz);
  dnorm[r] = nrm;
  float inv = 1.0f / nrm;
  float v[3] = {dx * inv, dy * inv, dz * inv};
  half_t* e = dirs_e + (size_t)r * 32;
  e[0] = (half_t)v[0]; e[1] = (half_t)v[1]; e[2] = (half_t)v[2];
  #pragma unroll
  for (int l = 0; l < 4; ++l) {
    float f = (float)(1 << l);
    #pragma unroll
    for (int c = 0; c < 3; ++c) {
      e[3 + l * 6 + c]     = (half_t)__sinf(v[c] * f);
      e[3 + l * 6 + 3 + c] = (half_t)__cosf(v[c] * f);
    }
  }
  for (int j = 27; j < 32; ++j) e[j] = (half_t)0.0f;
  float nr = nearp[r], fr = farp[r];
  for (int i = 0; i < NS_C; ++i) {
    float t = (float)i * (1.0f / (NS_C - 1));
    zc[r * NS_C + i] = nr * (1.0f - t) + fr * t;
  }
}

// ---------------------------------------------------------------------------
// Point embedding: pts = o + d*z ; embed L=10 -> 63 cols, f16 [P,64]
// ---------------------------------------------------------------------------
__global__ void pts_embed(const float* __restrict__ ro,
                          const float* __restrict__ rd,
                          const float* __restrict__ Z, int S,
                          half_t* __restrict__ E, int P) {
  int p = blockIdx.x * blockDim.x + threadIdx.x;
  if (p >= P) return;
  int r = p / S;
  float z = Z[p];
  float pt[3];
  #pragma unroll
  for (int c = 0; c < 3; ++c) pt[c] = ro[r * 3 + c] + rd[r * 3 + c] * z;
  half_t* e = E + (size_t)p * 64;
  e[0] = (half_t)pt[0]; e[1] = (half_t)pt[1]; e[2] = (half_t)pt[2];
  #pragma unroll
  for (int l = 0; l < 10; ++l) {
    float f = (float)(1 << l);
    #pragma unroll
    for (int c = 0; c < 3; ++c) {
      e[3 + l * 6 + c]     = (half_t)__sinf(pt[c] * f);
      e[3 + l * 6 + 3 + c] = (half_t)__cosf(pt[c] * f);
    }
  }
  e[63] = (half_t)0.0f;
}

// ---------------------------------------------------------------------------
// WMMA GEMM layer: C[P,N] = act(A[P,K] @ W + b), f16 in, f32 acc, f16 out.
// Block = 256 thr = 8 waves. Block tile = 64 rows x 128 cols.
// Wave w owns N-tile (blockIdx.y*128 + w*16) and iterates 4 M-tiles,
// reusing each B fragment across 4 WMMAs. A tile (64x32 f16 = 4KB) is
// double-buffered in LDS; next K-slab global load overlaps current WMMAs.
// All 4 A fragments + B fragment are materialized BEFORE the WMMAs so the
// compiler can stagger s_wait_dscnt (partial waits) instead of draining LDS
// before every matrix op. Fragment layouts per CDNA5 ISA 7.12.2; 4-dword
// groups loaded as aligned int4 -> ds_load_b128 / global_load_b128.
// ---------------------------------------------------------------------------
__global__ void __launch_bounds__(256) wmma_layer(
    const half_t* __restrict__ A, int lda, int K,
    const half_t* __restrict__ WT, int ldw,
    const float* __restrict__ bias,
    half_t* __restrict__ C, int ldc, int do_relu) {
  __shared__ int4 As[2][256];  // [buf][row*4 + quarter] : 64 rows x 32 halves
  const int tid  = threadIdx.x;
  const int lane = tid & 31;
  const int wave = tid >> 5;
  const int row0 = blockIdx.x * 64;
  const int n    = blockIdx.y * 128 + wave * 16 + (lane & 15);
  const int g    = lane >> 4;   // lane group (K-half select)
  const int m    = lane & 15;   // row within 16-row M-tile

  // staging: thread t loads int4 (8 halves) of row (t>>2), quarter (t&3)
  const int4* ag  = (const int4*)(A + (size_t)(row0 + (tid >> 2)) * lda) + (tid & 3);
  const int4* wp4 = (const int4*)(WT + (size_t)n * ldw);

  v8f zero = {};
  v8f c0 = zero, c1 = zero, c2 = zero, c3 = zero;
  union F { v16h v; int4 q[2]; };

  // prologue: stage first K-slab
  As[0][tid] = ag[0];
  ag += 4;
  int buf = 0;
  const int ksteps = K >> 5;
  for (int ks = 0; ks < ksteps; ++ks) {
    __syncthreads();
    const bool more = (ks + 1) < ksteps;
    int4 nxt;
    if (more) {
      nxt = ag[0];                 // overlap next A slab with WMMAs
      ag += 4;
      __builtin_prefetch(wp4 + 4, 0, 1);  // next weight slab (L2-hot)
    }
    // issue longest-latency loads first: B fragment from global (L2-hot)
    F b;
    b.q[0] = wp4[g];
    b.q[1] = wp4[2 + g];
    wp4 += 4;
    // materialize ALL A fragments before any WMMA -> staggered dscnt waits
    const int4* ab = As[buf];
    F a0, a1, a2, a3;
    a0.q[0] = ab[(m)*4 + g];        a0.q[1] = ab[(m)*4 + 2 + g];
    a1.q[0] = ab[(16 + m)*4 + g];   a1.q[1] = ab[(16 + m)*4 + 2 + g];
    a2.q[0] = ab[(32 + m)*4 + g];   a2.q[1] = ab[(32 + m)*4 + 2 + g];
    a3.q[0] = ab[(48 + m)*4 + g];   a3.q[1] = ab[(48 + m)*4 + 2 + g];
    c0 = __builtin_amdgcn_wmma_f32_16x16x32_f16(false, a0.v, false, b.v, (short)0, c0, false, false);
    c1 = __builtin_amdgcn_wmma_f32_16x16x32_f16(false, a1.v, false, b.v, (short)0, c1, false, false);
    c2 = __builtin_amdgcn_wmma_f32_16x16x32_f16(false, a2.v, false, b.v, (short)0, c2, false, false);
    c3 = __builtin_amdgcn_wmma_f32_16x16x32_f16(false, a3.v, false, b.v, (short)0, c3, false, false);
    if (more) As[buf ^ 1][tid] = nxt;
    buf ^= 1;
  }

  const float bn = bias[n];
  v8f acc[4] = {c0, c1, c2, c3};
  #pragma unroll
  for (int mt = 0; mt < 4; ++mt) {
    #pragma unroll
    for (int r8 = 0; r8 < 8; ++r8) {
      float val = acc[mt][r8] + bn;
      if (do_relu) val = fmaxf(val, 0.0f);
      // C/D layout: VGPR r8 -> row M = r8 + 8*g ; col = lane%16
      C[(size_t)(row0 + mt * 16 + r8 + g * 8) * ldc + n] = (half_t)val;
    }
  }
}

// ---------------------------------------------------------------------------
// Skip concat: O[p, 0:63]=pts_e, O[p, 63:319]=h, O[p,319]=0 (stride 320)
// ---------------------------------------------------------------------------
__global__ void concat_skip(const half_t* __restrict__ E,
                            const half_t* __restrict__ H,
                            half_t* __restrict__ O, int P) {
  int idx = blockIdx.x * blockDim.x + threadIdx.x;
  if (idx >= P * 320) return;
  int p = idx / 320, cc = idx - p * 320;
  half_t v;
  if (cc < 63)       v = E[(size_t)p * 64 + cc];
  else if (cc < 319) v = H[(size_t)p * 320 + (cc - 63)];
  else               v = (half_t)0.0f;
  O[(size_t)p * 320 + cc] = v;
}

// Fill view-input cols 256..319 with broadcast dir embedding (padded)
__global__ void fill_dirs(half_t* __restrict__ H,
                          const half_t* __restrict__ de, int S, int P) {
  int idx = blockIdx.x * blockDim.x + threadIdx.x;
  if (idx >= P * 64) return;
  int p = idx / 64, j = idx - p * 64;
  int ray = p / S;
  half_t v = (j < 32) ? de[(size_t)ray * 32 + j] : (half_t)0.0f;
  H[(size_t)p * 320 + 256 + j] = v;
}

// Tiny heads (alpha N=1, rgb N=3): plain GEMV, fp32 weights as given
__global__ void gemv_small(const half_t* __restrict__ A, int lda, int K,
                           const float* __restrict__ w,
                           const float* __restrict__ b,
                           float* __restrict__ out, int nout, int P) {
  int idx = blockIdx.x * blockDim.x + threadIdx.x;
  if (idx >= P * nout) return;
  int p = idx / nout, j = idx - p * nout;
  const half_t* ar = A + (size_t)p * lda;
  float acc = b[j];
  for (int k = 0; k < K; ++k) acc += (float)ar[k] * w[k * nout + j];
  out[(size_t)p * nout + j] = acc;
}

// ---------------------------------------------------------------------------
// Coarse volume rendering + deterministic inverse-CDF sampling + merge-sort
// ---------------------------------------------------------------------------
__global__ void render_coarse(const float* __restrict__ zc,
                              const float* __restrict__ araw,
                              const float* __restrict__ dnorm,
                              float* __restrict__ z_all) {
  int r = blockIdx.x * blockDim.x + threadIdx.x;
  if (r >= NRAYS) return;
  float z[NS_C], wts[NS_C];
  for (int i = 0; i < NS_C; ++i) z[i] = zc[r * NS_C + i];
  float nrm = dnorm[r], T = 1.0f;
  for (int i = 0; i < NS_C; ++i) {
    float d = ((i < NS_C - 1) ? (z[i + 1] - z[i]) : 1e10f) * nrm;
    float a = fmaxf(araw[r * NS_C + i], 0.0f);
    float alp = 1.0f - __expf(-a * d);
    wts[i] = alp * T;
    T *= (1.0f - alp + 1e-10f);
  }
  // pdf over weights[1:-1] (62 entries), cdf length 63 with leading 0
  float wv[62], cdf[63], bins[63];
  float wsum = 0.0f;
  for (int j = 0; j < 62; ++j) { wv[j] = wts[1 + j] + 1e-5f; wsum += wv[j]; }
  float inv = 1.0f / wsum;
  cdf[0] = 0.0f;
  for (int j = 0; j < 62; ++j) cdf[j + 1] = cdf[j] + wv[j] * inv;
  for (int i = 0; i < 63; ++i) bins[i] = 0.5f * (z[i] + z[i + 1]);
  float zs[128];
  for (int s = 0; s < 128; ++s) {
    float u = (float)s * (1.0f / 127.0f);
    int ind = 0;
    while (ind < 63 && cdf[ind] <= u) ++ind;   // searchsorted right
    int below = (ind - 1 < 0) ? 0 : ind - 1;
    int above = (ind > 62) ? 62 : ind;
    float den = cdf[above] - cdf[below];
    if (den < 1e-5f) den = 1.0f;
    float t = (u - cdf[below]) / den;
    zs[s] = bins[below] + t * (bins[above] - bins[below]);
  }
  // merge two sorted lists (z ascending, zs nondecreasing) -> z_all[192]
  float* za = z_all + (size_t)r * NS_F;
  int i = 0, j = 0;
  for (int k = 0; k < NS_F; ++k) {
    bool takeA = (j >= 128) || (i < NS_C && z[i] <= zs[j]);
    za[k] = takeA ? z[i++] : zs[j++];
  }
}

__global__ void render_fine(const float* __restrict__ z_all,
                            const float* __restrict__ araw,
                            const float* __restrict__ rraw,
                            const float* __restrict__ dnorm,
                            float* __restrict__ out) {
  int r = blockIdx.x * blockDim.x + threadIdx.x;
  if (r >= NRAYS) return;
  const float* z = z_all + (size_t)r * NS_F;
  float nrm = dnorm[r], T = 1.0f;
  float acc0 = 0.0f, acc1 = 0.0f, acc2 = 0.0f;
  for (int i = 0; i < NS_F; ++i) {
    float d = ((i < NS_F - 1) ? (z[i + 1] - z[i]) : 1e10f) * nrm;
    float a = fmaxf(araw[r * NS_F + i], 0.0f);
    float alp = 1.0f - __expf(-a * d);
    float wgt = alp * T;
    T *= (1.0f - alp + 1e-10f);
    const float* rr = rraw + (size_t)(r * NS_F + i) * 3;
    acc0 += wgt / (1.0f + __expf(-rr[0]));
    acc1 += wgt / (1.0f + __expf(-rr[1]));
    acc2 += wgt / (1.0f + __expf(-rr[2]));
  }
  out[r * 3 + 0] = acc0; out[r * 3 + 1] = acc1; out[r * 3 + 2] = acc2;
}

// ---------------------------------------------------------------------------
// Host orchestration
// ---------------------------------------------------------------------------
extern "C" void kernel_launch(void* const* d_in, const int* in_sizes, int n_in,
                              void* d_out, int out_size, void* d_ws, size_t ws_size,
                              hipStream_t stream) {
  (void)in_sizes; (void)n_in; (void)out_size; (void)ws_size;
  const float* rays_o = (const float*)d_in[0];
  const float* rays_d = (const float*)d_in[1];
  const float* nearp  = (const float*)d_in[2];
  const float* farp   = (const float*)d_in[3];
  void* const* prm_c = (void* const*)(d_in + 4);
  void* const* prm_f = (void* const*)(d_in + 28);

  // ---- workspace carve ----
  char* ws = (char*)d_ws;
  size_t off = 0;
  auto carve = [&](size_t bytes) -> void* {
    void* p = ws + off;
    off = (off + bytes + 255) & ~(size_t)255;
    return p;
  };
  half_t* E     = (half_t*)carve((size_t)PF * 64 * sizeof(half_t));
  half_t* HA    = (half_t*)carve((size_t)PF * 320 * sizeof(half_t));
  half_t* HB    = (half_t*)carve((size_t)PF * 320 * sizeof(half_t));
  half_t* de    = (half_t*)carve((size_t)NRAYS * 32 * sizeof(half_t));
  float*  dnorm = (float*)carve((size_t)NRAYS * sizeof(float));
  float*  zc    = (float*)carve((size_t)NRAYS * NS_C * sizeof(float));
  float*  za    = (float*)carve((size_t)NRAYS * NS_F * sizeof(float));
  float*  araw  = (float*)carve((size_t)PF * sizeof(float));
  float*  rraw  = (float*)carve((size_t)PF * 3 * sizeof(float));

  // GEMM layers: {param index of weight within net, K, N, Kpad}
  struct LD { int widx, K, N, Kpad; };
  const LD defs[10] = {
      {0, 63, 256, 64},   {2, 256, 256, 256}, {4, 256, 256, 256},
      {6, 256, 256, 256}, {8, 256, 256, 256}, {10, 319, 256, 320},
      {12, 256, 256, 256}, {14, 256, 256, 256},
      {16, 256, 256, 256},  // w_feat
      {20, 283, 128, 288},  // w_view
  };
  half_t* wt[2][10];
  for (int net = 0; net < 2; ++net) {
    void* const* prm = net ? prm_f : prm_c;
    for (int l = 0; l < 10; ++l) {
      size_t elems = (size_t)defs[l].N * defs[l].Kpad;
      wt[net][l] = (half_t*)carve(elems * sizeof(half_t));
      int threads = (int)elems;
      prep_wt<<<(threads + 255) / 256, 256, 0, stream>>>(
          (const float*)prm[defs[l].widx], wt[net][l],
          defs[l].K, defs[l].N, defs[l].Kpad);
    }
  }

  ray_prep<<<(NRAYS + 255) / 256, 256, 0, stream>>>(rays_d, nearp, farp,
                                                    de, dnorm, zc);

  auto run_pass = [&](void* const* prm, half_t** w, const float* Z, int S,
                      bool fine) {
    const int P = NRAYS * S;     // multiple of 64 (32768 / 98304)
    pts_embed<<<(P + 255) / 256, 256, 0, stream>>>(rays_o, rays_d, Z, S, E, P);

    half_t* X = HA; half_t* Y = HB;
    // L0: E -> X
    wmma_layer<<<dim3(P / 64, 2), 256, 0, stream>>>(
        E, 64, 64, w[0], 64, (const float*)prm[1], X, 320, 1);
    // L1..L4
    for (int i = 1; i <= 4; ++i) {
      wmma_layer<<<dim3(P / 64, 2), 256, 0, stream>>>(
          X, 320, 256, w[i], 256, (const float*)prm[2 * i + 1], Y, 320, 1);
      half_t* t = X; X = Y; Y = t;
    }
    // skip concat: [pts_e | h] -> Y (320 wide)
    concat_skip<<<((size_t)P * 320 + 255) / 256, 256, 0, stream>>>(E, X, Y, P);
    // L5 (K=320): Y -> X
    wmma_layer<<<dim3(P / 64, 2), 256, 0, stream>>>(
        Y, 320, 320, w[5], 320, (const float*)prm[11], X, 320, 1);
    // L6: X -> Y ; L7: Y -> X
    wmma_layer<<<dim3(P / 64, 2), 256, 0, stream>>>(
        X, 320, 256, w[6], 256, (const float*)prm[13], Y, 320, 1);
    wmma_layer<<<dim3(P / 64, 2), 256, 0, stream>>>(
        Y, 320, 256, w[7], 256, (const float*)prm[15], X, 320, 1);
    // alpha head (w_alpha=prm[18], b_alpha=prm[19])
    gemv_small<<<(P + 255) / 256, 256, 0, stream>>>(
        X, 320, 256, (const float*)prm[18], (const float*)prm[19], araw, 1, P);
    if (fine) {
      // feat (no relu): X -> Y cols 0..255
      wmma_layer<<<dim3(P / 64, 2), 256, 0, stream>>>(
          X, 320, 256, w[8], 256, (const float*)prm[17], Y, 320, 0);
      // append dir embedding into cols 256..319 of Y
      fill_dirs<<<((size_t)P * 64 + 255) / 256, 256, 0, stream>>>(Y, de, S, P);
      // view layer (K=288, N=128, relu): Y -> X
      wmma_layer<<<dim3(P / 64, 1), 256, 0, stream>>>(
          Y, 320, 288, w[9], 288, (const float*)prm[21], X, 320, 1);
      // rgb head (w_rgb=prm[22], b_rgb=prm[23])
      gemv_small<<<((size_t)P * 3 + 255) / 256, 256, 0, stream>>>(
          X, 320, 128, (const float*)prm[22], (const float*)prm[23], rraw, 3, P);
    }
  };

  // coarse pass (alpha only) -> hierarchical sampling -> z_all
  run_pass(prm_c, wt[0], zc, NS_C, false);
  render_coarse<<<(NRAYS + 63) / 64, 64, 0, stream>>>(zc, araw, dnorm, za);

  // fine pass -> final rgb
  run_pass(prm_f, wt[1], za, NS_F, true);
  render_fine<<<(NRAYS + 63) / 64, 64, 0, stream>>>(za, araw, rraw, dnorm,
                                                    (float*)d_out);
}
